// EnergyLSTM_876173328995
// MI455X (gfx1250) — compile-verified, matrix operand order checked by reading, not run on hardware
//
#include <hip/hip_runtime.h>

typedef __attribute__((ext_vector_type(16))) _Float16 v16h;
typedef __attribute__((ext_vector_type(8)))  float    v8f;

#define B_TOT 4096
#define T_LEN 512
#define HID   32

__device__ __forceinline__ float fsigmoid(float x) {
    // 1/(1+exp(-x)) via hardware exp2 + rcp
    return __builtin_amdgcn_rcpf(1.0f + __builtin_amdgcn_exp2f(x * -1.44269504088896340736f));
}
__device__ __forceinline__ float ftanh_fast(float x) {
    // tanh(x) = 2*sigmoid(2x) - 1
    return 2.0f * __builtin_amdgcn_rcpf(1.0f + __builtin_amdgcn_exp2f(x * -2.88539008177792681472f)) - 1.0f;
}

// Pre-pass: transpose input_seq [B,T] -> xT [T,B] so the per-step batch read is coalesced.
__global__ void transpose_x_kernel(const float* __restrict__ x, float* __restrict__ xT) {
    __shared__ float tile[32][33];
    const int t0 = blockIdx.x * 32;
    const int b0 = blockIdx.y * 32;
    const int tx = threadIdx.x;      // 0..31
    const int ty = threadIdx.y;      // 0..7
    for (int i = ty; i < 32; i += 8)
        tile[i][tx] = x[(size_t)(b0 + i) * T_LEN + (t0 + tx)];
    __syncthreads();
    for (int i = ty; i < 32; i += 8)
        xT[(size_t)(t0 + i) * B_TOT + (b0 + tx)] = tile[tx][i];
}

// One workgroup (2 waves, 64 threads) handles 16 batch rows for all 512 steps.
// Wave w owns hidden columns [16w, 16w+16): its 4 gate tiles (i,f,g,o) are one
// v_wmma_f32_16x16x32_f16 each (K=32 == H consumed in a single instruction).
__global__ __launch_bounds__(64) void lstm_wmma_kernel(
    const float* __restrict__ xsrc, int x_transposed,
    const float* __restrict__ W_ih, const float* __restrict__ W_hh,
    const float* __restrict__ b_ih, const float* __restrict__ b_hh,
    const float* __restrict__ W_out, const float* __restrict__ b_out,
    float* __restrict__ out)
{
    __shared__ __align__(16) _Float16 hbuf[2][16][HID]; // double-buffered h, row-major [row][k]
    __shared__ float out_acc[16];

    const int tid  = threadIdx.x;   // 0..63
    const int wave = tid >> 5;      // 0..1 : hidden-column half
    const int ln   = tid & 31;      // lane in wave
    const int g    = ln >> 4;       // lane group (0/1)
    const int nl   = ln & 15;       // lane within group
    const int b0   = blockIdx.x * 16;

    // ---- Resident B-matrices: W_hh^T tiles for this wave's 4 gates, f16, ISA B-layout.
    // B[K=k, N=nl] of tile (gate G, half `wave`) = W_hh[(G*32 + wave*16 + nl)*32 + k]
    // element e of v16h at lane group g -> K = (e&7) + 8*( 2*(e>>3) + g )
    v16h Bm[4];
    for (int G = 0; G < 4; ++G) {
        const int gcol = G * HID + wave * 16 + nl;
        for (int e = 0; e < 16; ++e) {
            const int k = (e & 7) + 8 * (((e >> 3) << 1) + g);
            Bm[G][e] = (_Float16)W_hh[gcol * HID + k];
        }
    }
    // Per-lane input weight + combined bias for each gate tile (I == 1).
    float wi[4], bsum[4];
    for (int G = 0; G < 4; ++G) {
        const int col = G * HID + wave * 16 + nl;
        wi[G]   = W_ih[col];
        bsum[G] = b_ih[col] + b_hh[col];
    }

    // ---- init: h = 0 (buffer 0), c = 0 (registers)
    float c[8];
#pragma unroll
    for (int r = 0; r < 8; ++r) c[r] = 0.0f;
    for (int i = tid; i < 16 * HID; i += 64) hbuf[0][0][i] = (_Float16)0.0f;
    if (tid < 16) out_acc[tid] = 0.0f;
    __syncthreads();

    float hout[8]; // last-step h (f32) for the final projection

    for (int t = 0; t < T_LEN; ++t) {
        const int cur = t & 1;

        // A = h (16x32 f16) in A-layout: lane -> row nl; elements 0..7 hold
        // K = 8g..8g+7, elements 8..15 hold K = 16+8g..23+8g (K-contiguous runs
        // of 8 -> two ds_load_b128 from the row-major LDS tile).
        v16h A;
        {
            const _Float16* hp = &hbuf[cur][nl][0];
            union { uint4 u[2]; v16h v; } cvt;
            cvt.u[0] = *(const uint4*)(hp + 8 * g);
            cvt.u[1] = *(const uint4*)(hp + 16 + 8 * g);
            A = cvt.v;
        }

        // x[t] for the 16 batch rows: one coalesced load + lane broadcasts.
        float xw;
        if (x_transposed) xw = xsrc[(size_t)t * B_TOT + (b0 + nl)];
        else              xw = xsrc[(size_t)(b0 + nl) * T_LEN + t];
        float xrow[8];
#pragma unroll
        for (int r = 0; r < 8; ++r) xrow[r] = __shfl(xw, r + 8 * g, 32);

        // 4 WMMAs: D = h @ W_hh_tile^T + (x*W_ih + b) folded into C.
        float gate[4][8];
#pragma unroll
        for (int G = 0; G < 4; ++G) {
            v8f Cv;
#pragma unroll
            for (int r = 0; r < 8; ++r) Cv[r] = xrow[r] * wi[G] + bsum[G];
            v8f D = __builtin_amdgcn_wmma_f32_16x16x32_f16(
                false, A, false, Bm[G], (short)0, Cv, false, false);
#pragma unroll
            for (int r = 0; r < 8; ++r) gate[G][r] = D[r];
        }

        // Elementwise LSTM cell. This lane owns hidden column wave*16+nl for
        // rows r + 8g (C/D layout), matching c[] residency exactly.
        const int mycol = wave * 16 + nl;
#pragma unroll
        for (int r = 0; r < 8; ++r) {
            const float ig = fsigmoid(gate[0][r]);
            const float fg = fsigmoid(gate[1][r]);
            const float gg = ftanh_fast(gate[2][r]);
            const float og = fsigmoid(gate[3][r]);
            c[r] = fg * c[r] + ig * gg;
            const float h = og * ftanh_fast(c[r]);
            hout[r] = h;
            hbuf[cur ^ 1][r + 8 * g][mycol] = (_Float16)h;   // next step's A source
        }
        __syncthreads(); // double buffer -> single barrier per step
    }

    // ---- final projection: out[b] = sum_k h_T[b,k] * W_out[k] + b_out
    {
        const float wo = W_out[wave * 16 + nl];
#pragma unroll
        for (int r = 0; r < 8; ++r)
            atomicAdd(&out_acc[r + 8 * g], hout[r] * wo);
    }
    __syncthreads();
    if (tid < 16) out[b0 + tid] = out_acc[tid] + b_out[0];
}

extern "C" void kernel_launch(void* const* d_in, const int* in_sizes, int n_in,
                              void* d_out, int out_size, void* d_ws, size_t ws_size,
                              hipStream_t stream) {
    const float* x     = (const float*)d_in[0];
    const float* W_ih  = (const float*)d_in[1];
    const float* W_hh  = (const float*)d_in[2];
    const float* b_ih  = (const float*)d_in[3];
    const float* b_hh  = (const float*)d_in[4];
    const float* W_out = (const float*)d_in[5];
    const float* b_out = (const float*)d_in[6];
    float* out = (float*)d_out;

    const size_t xbytes = (size_t)B_TOT * T_LEN * sizeof(float);
    int use_tr = (ws_size >= xbytes) ? 1 : 0;
    const float* xsrc = x;
    if (use_tr) {
        float* xT = (float*)d_ws;
        dim3 grid(T_LEN / 32, B_TOT / 32);
        transpose_x_kernel<<<grid, dim3(32, 8), 0, stream>>>(x, xT);
        xsrc = xT;
    }
    lstm_wmma_kernel<<<B_TOT / 16, 64, 0, stream>>>(
        xsrc, use_tr, W_ih, W_hh, b_ih, b_hh, W_out, b_out, out);
}
